// SelfAttention_64476049047717
// MI455X (gfx1250) — matrix operand order repaired and checked by reading, not compiled, not run on hardware
//
#include <hip/hip_runtime.h>

typedef __attribute__((ext_vector_type(16))) __bf16 v16bf;
typedef __attribute__((ext_vector_type(8)))  __bf16 v8bf;
typedef __attribute__((ext_vector_type(4)))  __bf16 v4bf;
typedef __attribute__((ext_vector_type(2)))  __bf16 v2bf;
typedef __attribute__((ext_vector_type(8)))  float  v8f;

constexpr int Bb = 4;      // batch
constexpr int Ss = 4096;   // sequence
constexpr int Dd = 64;     // head dim
constexpr int BM = 64;     // query rows per workgroup (4 waves x 16)
constexpr int BN = 64;     // keys per KV block

__device__ __forceinline__ __bf16 f2bf(float f) { return (__bf16)f; }

// D = A(16x32) x B(32x16) + C, bf16 in / f32 accumulate
__device__ __forceinline__ v8f wmma_bf16(v16bf a, v16bf b, v8f c) {
    return __builtin_amdgcn_wmma_f32_16x16x32_bf16(false, a, false, b,
                                                   (short)0, c, false, false);
}

__global__ __launch_bounds__(128)
void fa_fwd_kernel(const float* __restrict__ Q, const float* __restrict__ K,
                   const float* __restrict__ V, float* __restrict__ O)
{
    __shared__ alignas(16) __bf16 ldsK [BN * Dd];   // K block, row-major [key][d]
    __shared__ alignas(16) __bf16 ldsVT[Dd * BN];   // V block transposed [d][key]

    const int tid  = threadIdx.x;
    const int w    = tid >> 5;        // wave id 0..3
    const int lane = tid & 31;
    const int ln   = lane & 15;       // lane within half
    const int h    = lane >> 4;       // lane half (0/1)

    const int mblk = blockIdx.x & (Ss / BM - 1);
    const int b    = blockIdx.x / (Ss / BM);
    const int m0   = mblk * BM;

    const float* Qb = Q + (size_t)b * Ss * Dd;
    const float* Kb = K + (size_t)b * Ss * Dd;
    const float* Vb = V + (size_t)b * Ss * Dd;
    float*       Ob = O + (size_t)b * Ss * Dd;

    // ---- Q as B-fragments of Q^T (lane = query column); fold 1/sqrt(D) * log2(e)
    const float scale = 0.125f * 1.44269504088896340736f;
    const int   qg    = m0 + w * 16 + ln;           // this lane's query row
    v16bf qb[2];
    #pragma unroll
    for (int c = 0; c < 2; ++c) {
        const float* qp = Qb + (size_t)qg * Dd + c * 32 + h * 8;
        #pragma unroll
        for (int e = 0; e < 16; ++e)                // B-frag elem e <-> d = c*32 + (e>>3)*16 + h*8 + (e&7)
            qb[c][e] = f2bf(qp[e + (e >> 3) * 8] * scale);
    }

    // O in normal C-layout: elem r <-> query row h*8+r (in-wave), col d = dt*16 + ln
    v8f oacc[4] = {};
    float m_i = -1e30f, l_i = 0.0f;   // per-query stats (query = ln, one scalar per lane)

    for (int j = 0; j <= mblk; ++j) {
        const int kv0 = j * BN;

        __syncthreads();     // protect LDS WAR from previous iteration
        // ---- cooperative stage: K -> ldsK bf16, V -> ldsVT bf16 (row pairs packed)
        #pragma unroll
        for (int i = 0; i < 4; ++i) {
            const int p   = tid + i * 128;          // 512 row-pair x col4 units
            const int row = (p >> 4) * 2;
            const int col = (p & 15) * 4;
            const float* kp = Kb + (size_t)(kv0 + row) * Dd + col;
            const float* vp = Vb + (size_t)(kv0 + row) * Dd + col;
            const float4 k0 = *(const float4*)kp;
            const float4 k1 = *(const float4*)(kp + Dd);
            const float4 v0 = *(const float4*)vp;
            const float4 v1 = *(const float4*)(vp + Dd);
            v4bf kb0; kb0[0]=f2bf(k0.x); kb0[1]=f2bf(k0.y); kb0[2]=f2bf(k0.z); kb0[3]=f2bf(k0.w);
            v4bf kb1; kb1[0]=f2bf(k1.x); kb1[1]=f2bf(k1.y); kb1[2]=f2bf(k1.z); kb1[3]=f2bf(k1.w);
            *(v4bf*)&ldsK[(row    ) * Dd + col] = kb0;
            *(v4bf*)&ldsK[(row + 1) * Dd + col] = kb1;
            v2bf t0; t0[0]=f2bf(v0.x); t0[1]=f2bf(v1.x);
            v2bf t1; t1[0]=f2bf(v0.y); t1[1]=f2bf(v1.y);
            v2bf t2; t2[0]=f2bf(v0.z); t2[1]=f2bf(v1.z);
            v2bf t3; t3[0]=f2bf(v0.w); t3[1]=f2bf(v1.w);
            *(v2bf*)&ldsVT[(col + 0) * BN + row] = t0;
            *(v2bf*)&ldsVT[(col + 1) * BN + row] = t1;
            *(v2bf*)&ldsVT[(col + 2) * BN + row] = t2;
            *(v2bf*)&ldsVT[(col + 3) * BN + row] = t3;
        }
        __syncthreads();

        // ---- S^T = K Q^T : 4 key-tiles, A = K tile (lane = key row)
        v8f st[4];
        #pragma unroll
        for (int t = 0; t < 4; ++t) {
            const __bf16* kr = &ldsK[(t * 16 + ln) * Dd + h * 8];
            v8bf a00 = *(const v8bf*)&kr[ 0];       // e0..7  : d = h*8 + e
            v8bf a01 = *(const v8bf*)&kr[16];       // e8..15 : d = 16 + h*8 + e
            v8bf a10 = *(const v8bf*)&kr[32];
            v8bf a11 = *(const v8bf*)&kr[48];
            v16bf ka0 = __builtin_shufflevector(a00, a01, 0,1,2,3,4,5,6,7,8,9,10,11,12,13,14,15);
            v16bf ka1 = __builtin_shufflevector(a10, a11, 0,1,2,3,4,5,6,7,8,9,10,11,12,13,14,15);
            v8f acc = {};
            acc = wmma_bf16(ka0, qb[0], acc);
            acc = wmma_bf16(ka1, qb[1], acc);
            st[t] = acc;                            // elem r <-> key = t*16 + h*8 + r
        }

        // ---- causal mask (diagonal block only)
        if (j == mblk) {
            #pragma unroll
            for (int t = 0; t < 4; ++t)
                #pragma unroll
                for (int r = 0; r < 8; ++r)
                    if (kv0 + t * 16 + h * 8 + r > qg) st[t][r] = -1e30f;
        }

        // ---- online softmax in log2 domain: one query per lane
        float mx = st[0][0];
        #pragma unroll
        for (int t = 0; t < 4; ++t)
            #pragma unroll
            for (int r = 0; r < 8; ++r) mx = fmaxf(mx, st[t][r]);
        mx = fmaxf(mx, __shfl_xor(mx, 16, 32));     // other half holds other 32 keys
        const float mnew  = fmaxf(m_i, mx);
        const float alpha = __builtin_amdgcn_exp2f(m_i - mnew);
        m_i = mnew;

        float sum = 0.0f;
        #pragma unroll
        for (int t = 0; t < 4; ++t)
            #pragma unroll
            for (int r = 0; r < 8; ++r) {
                const float p = __builtin_amdgcn_exp2f(st[t][r] - mnew);
                st[t][r] = p;
                sum += p;
            }
        sum += __shfl_xor(sum, 16, 32);
        l_i = l_i * alpha + sum;

        // ---- broadcast alpha to O's row layout (query h*8+r lives on lane h*8+r)
        float alpha_b[8];
        #pragma unroll
        for (int r = 0; r < 8; ++r) alpha_b[r] = __shfl(alpha, h * 8 + r, 32);
        #pragma unroll
        for (int dt = 0; dt < 4; ++dt)
            #pragma unroll
            for (int r = 0; r < 8; ++r) oacc[dt][r] *= alpha_b[r];

        // ---- P as A-fragments: lane's own st values, no cross-lane traffic at all.
        // A elem e, chunk c: key = c*32 + (e>>3)*16 + h*8 + (e&7)
        //   e=0..7  -> st[2c  ][e]  (tile 2c,   rows h*8+e)
        //   e=8..15 -> st[2c+1][e-8](tile 2c+1, rows h*8+e-8)
        v16bf pa[2];
        #pragma unroll
        for (int c = 0; c < 2; ++c) {
            #pragma unroll
            for (int e = 0; e < 8; ++e) {
                pa[c][e]     = f2bf(st[2*c    ][e]);
                pa[c][e + 8] = f2bf(st[2*c + 1][e]);
            }
        }

        // ---- O += P V : 4 d-tiles, B = V (lane = d column, contiguous keys from ldsVT)
        #pragma unroll
        for (int dt = 0; dt < 4; ++dt) {
            const __bf16* vb = &ldsVT[(dt * 16 + ln) * BN + h * 8];
            v16bf vb0 = __builtin_shufflevector(*(const v8bf*)&vb[ 0], *(const v8bf*)&vb[16], 0,1,2,3,4,5,6,7,8,9,10,11,12,13,14,15);     // keys  0 + (e>>3)*16 + h*8 + (e&7)
            v16bf vb1 = __builtin_shufflevector(*(const v8bf*)&vb[32], *(const v8bf*)&vb[48], 0,1,2,3,4,5,6,7,8,9,10,11,12,13,14,15);     // keys 32 + (e>>3)*16 + h*8 + (e&7)
            oacc[dt] = wmma_bf16(pa[0], vb0, oacc[dt]);
            oacc[dt] = wmma_bf16(pa[1], vb1, oacc[dt]);
        }
    }

    // ---- epilogue: broadcast 1/l to row layout, normalize, store
    const float inv = 1.0f / l_i;
    float inv_b[8];
    #pragma unroll
    for (int r = 0; r < 8; ++r) inv_b[r] = __shfl(inv, h * 8 + r, 32);
    #pragma unroll
    for (int r = 0; r < 8; ++r) {
        float* rp = Ob + (size_t)(m0 + w * 16 + h * 8 + r) * Dd + ln;
        #pragma unroll
        for (int dt = 0; dt < 4; ++dt)
            rp[dt * 16] = oacc[dt][r] * inv_b[r];
    }
}

extern "C" void kernel_launch(void* const* d_in, const int* in_sizes, int n_in,
                              void* d_out, int out_size, void* d_ws, size_t ws_size,
                              hipStream_t stream) {
    const float* q = (const float*)d_in[0];
    const float* k = (const float*)d_in[1];
    const float* v = (const float*)d_in[2];
    float* out = (float*)d_out;

    dim3 grid(Bb * (Ss / BM));   // 4 * 64 = 256 workgroups
    dim3 block(128);             // 4 wave32s
    hipLaunchKernelGGL(fa_fwd_kernel, grid, block, 0, stream, q, k, v, out);
}
